// Seq2Seq_28363964023389
// MI455X (gfx1250) — matrix operand order repaired and checked by reading, not compiled, not run on hardware
//
#include <hip/hip_runtime.h>
#include <hip/hip_bf16.h>
#include <stdint.h>

// Sizes from the reference
#define BV   32000   // vocab
#define EE   256     // embed
#define HH   512     // hidden
#define BB   32      // batch
#define TX   512     // encoder steps
#define TY   128
#define TDEC 127     // decoder steps
#define G3H  1536    // 3*H

typedef __attribute__((ext_vector_type(16))) __bf16 v16bf;
typedef __attribute__((ext_vector_type(8)))  __bf16 v8bf;
typedef __attribute__((ext_vector_type(8)))  float  v8f;

__device__ __forceinline__ uint16_t f2bf(float f) {
  uint32_t u = __float_as_uint(f);
  u += 0x7FFFu + ((u >> 16) & 1u);   // round-to-nearest-even
  return (uint16_t)(u >> 16);
}

// A fragment: 16x32 bf16, contiguous k order per lane half (same order as B).
// lane L holds M = L%16; lanes 0-15: K = k0+0..15; lanes 16-31: K = k0+16..31 (contiguous 32B)
__device__ __forceinline__ v16bf load_a_frag(const __bf16* A, int lda, int row0, int k0, int lane) {
  int m  = row0 + (lane & 15);
  int kb = k0 + ((lane >> 4) << 4);
  return *(const v16bf*)(A + (size_t)m * lda + kb);
}

// B fragment: 32x16 bf16, B[k][n] = W[n][k] (W row-major NxK -> computes A @ W^T).
// lane L holds N = L%16; lanes 0-15: K = k0+0..15; lanes 16-31: K = k0+16..31 (contiguous 32B)
__device__ __forceinline__ v16bf load_b_frag(const __bf16* W, int ldw, int col0, int k0, int lane) {
  int n  = col0 + (lane & 15);
  int kb = k0 + ((lane >> 4) << 4);
  return *(const v16bf*)(W + (size_t)n * ldw + kb);
}

// ---------------- fused gate GEMM: Gx = Ax@Wx^T, Gh = Ah@Wh^T (one wave = one 16x16 tile)
__global__ __launch_bounds__(256) void gates_gemm_kernel(
    const uint16_t* __restrict__ Ax_, int ldax, int Kx, const uint16_t* __restrict__ Wx_,
    const uint16_t* __restrict__ Ah_, int Kh,          const uint16_t* __restrict__ Wh_,
    float* __restrict__ Gx, float* __restrict__ Gh, int M, int N)
{
  int wave = (blockIdx.x * blockDim.x + threadIdx.x) >> 5;
  int lane = threadIdx.x & 31;
  int tiles_n = N >> 4;
  int tpm = (M >> 4) * tiles_n;
  if (wave >= 2 * tpm) return;                 // uniform per-wave: EXEC stays all-ones
  bool isH = wave >= tpm;
  int tl = isH ? wave - tpm : wave;
  int tm = tl / tiles_n, tn = tl % tiles_n;
  const __bf16* A = (const __bf16*)(isH ? Ah_ : Ax_);
  const __bf16* W = (const __bf16*)(isH ? Wh_ : Wx_);
  int K   = isH ? Kh : Kx;
  int lda = isH ? Kh : ldax;
  float* G = isH ? Gh : Gx;

  v8f c = {};
  for (int k0 = 0; k0 < K; k0 += 32) {
    v16bf a = load_a_frag(A, lda, tm << 4, k0, lane);
    v16bf b = load_b_frag(W, K,   tn << 4, k0, lane);
    c = __builtin_amdgcn_wmma_f32_16x16x32_bf16(false, a, false, b, (short)0, c, false, false);
  }
  int n  = (tn << 4) + (lane & 15);
  int mb = (tm << 4) + ((lane >> 4) << 3);
#pragma unroll
  for (int r = 0; r < 8; ++r)
    G[(size_t)(mb + r) * N + n] = c[r];
}

// ---------------- final FC: C = A@W^T + bias  (M=4064, N=32000, K=512)
// Register-blocked: each wave computes a 32x64 output block (2 M-tiles x 4 N-tiles),
// reusing A fragments across 4 N-tiles -> ~21 FLOP/byte of cache traffic.
__global__ __launch_bounds__(256) void gemm_bias_kernel(
    const uint16_t* __restrict__ A_, const uint16_t* __restrict__ W_,
    const float* __restrict__ bias, float* __restrict__ C, int M, int N, int K)
{
  int wave = (blockIdx.x * blockDim.x + threadIdx.x) >> 5;
  int lane = threadIdx.x & 31;
  int nblk = N >> 6;                                  // 64-wide N blocks
  int total = (M >> 5) * nblk;                        // 32-tall M blocks
  if (wave >= total) return;
  int bm = wave / nblk, bn = wave % nblk;
  const __bf16* A = (const __bf16*)A_;
  const __bf16* W = (const __bf16*)W_;
  int row0 = bm << 5;
  int col0 = bn << 6;

  v8f c[2][4];
#pragma unroll
  for (int i = 0; i < 2; ++i)
#pragma unroll
    for (int j = 0; j < 4; ++j) c[i][j] = (v8f){};

  for (int k0 = 0; k0 < K; k0 += 32) {
    v16bf a0 = load_a_frag(A, K, row0,      k0, lane);
    v16bf a1 = load_a_frag(A, K, row0 + 16, k0, lane);
#pragma unroll
    for (int j = 0; j < 4; ++j) {
      v16bf b = load_b_frag(W, K, col0 + (j << 4), k0, lane);
      c[0][j] = __builtin_amdgcn_wmma_f32_16x16x32_bf16(false, a0, false, b, (short)0, c[0][j], false, false);
      c[1][j] = __builtin_amdgcn_wmma_f32_16x16x32_bf16(false, a1, false, b, (short)0, c[1][j], false, false);
    }
  }

#pragma unroll
  for (int j = 0; j < 4; ++j) {
    int n = col0 + (j << 4) + (lane & 15);
    float bv = bias[n];
#pragma unroll
    for (int i = 0; i < 2; ++i) {
      int mb = row0 + (i << 4) + ((lane >> 4) << 3);
#pragma unroll
      for (int r = 0; r < 8; ++r)
        C[(size_t)(mb + r) * N + n] = c[i][j][r] + bv;
    }
  }
}

// ---------------- GRU elementwise update (shared by encoder & decoder via nullable args)
__global__ __launch_bounds__(256) void gru_update_kernel(
    const float* __restrict__ gx, const float* __restrict__ gh,
    const float* __restrict__ b_ih, const float* __restrict__ b_hh,
    float* __restrict__ h, uint16_t* __restrict__ h_bf,
    float* __restrict__ eh_out, long eh_stride,
    const float* __restrict__ sx, uint16_t* __restrict__ outs_row, long outs_stride,
    float* __restrict__ ox_copy)
{
  int gid = blockIdx.x * blockDim.x + threadIdx.x;
  if (gid >= BB * HH) return;
  int b = gid / HH, i = gid % HH;
  const float* gxb = gx + (size_t)b * G3H;
  const float* ghb = gh + (size_t)b * G3H;
  float rin = gxb[i]          + b_ih[i]          + ghb[i]          + b_hh[i];
  float zin = gxb[HH + i]     + b_ih[HH + i]     + ghb[HH + i]     + b_hh[HH + i];
  float r = 1.f / (1.f + __expf(-rin));
  float z = 1.f / (1.f + __expf(-zin));
  float gxn = gxb[2 * HH + i] + b_ih[2 * HH + i];
  float ghn = ghb[2 * HH + i] + b_hh[2 * HH + i];
  float nn = tanhf(gxn + r * ghn);
  float hnew = (1.f - z) * nn + z * h[gid];
  h[gid] = hnew;
  h_bf[gid] = f2bf(hnew);
  if (eh_out)  eh_out[(size_t)b * eh_stride + i] = hnew;
  if (ox_copy) ox_copy[gid] = hnew;
  if (outs_row) outs_row[(size_t)b * outs_stride + i] = f2bf(hnew + sx[gid]);
}

// ---------------- fused attention: scores -> softmax -> context -> build decoder input row
__global__ __launch_bounds__(512) void attn_kernel(
    const float* __restrict__ eh, const float* __restrict__ ox,
    const uint16_t* __restrict__ dec_emb, uint16_t* __restrict__ dec_in,
    float* __restrict__ sx_out, int t)
{
  __shared__ float sc[TX];
  __shared__ float red[TX];
  int b = blockIdx.x, tid = threadIdx.x;      // 512 threads: tid = time (scores) then h (context)
  const float* oxb = ox + (size_t)b * HH;
  const float* ehb = eh + (size_t)b * TX * HH;

  const float* row = ehb + (size_t)tid * HH;
  float s = 0.f;
#pragma unroll 8
  for (int k = 0; k < HH; ++k) s += row[k] * oxb[k];
  red[tid] = s;
  __syncthreads();
  for (int off = TX / 2; off > 0; off >>= 1) {
    if (tid < off) red[tid] = fmaxf(red[tid], red[tid + off]);
    __syncthreads();
  }
  float mx = red[0];
  __syncthreads();
  float e = __expf(s - mx);
  red[tid] = e;
  __syncthreads();
  for (int off = TX / 2; off > 0; off >>= 1) {
    if (tid < off) red[tid] += red[tid + off];
    __syncthreads();
  }
  float inv = 1.f / red[0];
  sc[tid] = e * inv;                           // attention weight a[b][tid]
  __syncthreads();

  float acc = 0.f;                             // context: sx[b][tid] = sum_t a[t]*eh[b][t][tid]
#pragma unroll 4
  for (int tt = 0; tt < TX; ++tt) acc += sc[tt] * ehb[(size_t)tt * HH + tid];
  sx_out[(size_t)b * HH + tid] = acc;
  dec_in[(size_t)b * (EE + HH) + EE + tid] = f2bf(acc);
  if (tid < EE)                                // concat([embedded token, sx])
    dec_in[(size_t)b * (EE + HH) + tid] = dec_emb[((size_t)b * TDEC + t) * EE + tid];
}

// ---------------- small helpers
__global__ void cvt_bf16_kernel(const float* __restrict__ s, uint16_t* __restrict__ d, int n) {
  int i = blockIdx.x * blockDim.x + threadIdx.x;
  if (i < n) d[i] = f2bf(s[i]);
}
__global__ void embed_kernel(const float* __restrict__ emb, const int* __restrict__ toks,
                             uint16_t* __restrict__ out, int Tsel, int Tstride, int n) {
  int gid = blockIdx.x * blockDim.x + threadIdx.x;
  if (gid >= n) return;
  int e = gid % EE; int bt = gid / EE; int b = bt / Tsel; int t = bt % Tsel;
  int tok = toks[b * Tstride + t];
  out[gid] = f2bf(emb[(size_t)tok * EE + e]);
}
__global__ void fill_zero_kernel(uint32_t* p, int n) {
  int i = blockIdx.x * blockDim.x + threadIdx.x;
  if (i < n) p[i] = 0u;
}
__global__ void init_ox_kernel(const float* __restrict__ h_init, float* __restrict__ ox) {
  int gid = blockIdx.x * blockDim.x + threadIdx.x;
  if (gid < BB * HH) ox[gid] = h_init[gid % HH];
}

extern "C" void kernel_launch(void* const* d_in, const int* in_sizes, int n_in,
                              void* d_out, int out_size, void* d_ws, size_t ws_size,
                              hipStream_t stream)
{
  const int*   x         = (const int*)  d_in[0];
  const int*   y         = (const int*)  d_in[1];
  const float* embedding = (const float*)d_in[2];
  const float* enc_W_ih  = (const float*)d_in[3];
  const float* enc_W_hh  = (const float*)d_in[4];
  const float* enc_b_ih  = (const float*)d_in[5];
  const float* enc_b_hh  = (const float*)d_in[6];
  const float* dec_W_ih  = (const float*)d_in[7];
  const float* dec_W_hh  = (const float*)d_in[8];
  const float* dec_b_ih  = (const float*)d_in[9];
  const float* dec_b_hh  = (const float*)d_in[10];
  const float* h_init    = (const float*)d_in[11];
  const float* fc_W      = (const float*)d_in[12];
  const float* fc_b      = (const float*)d_in[13];
  float* out = (float*)d_out;

  // bump allocator over d_ws (256B aligned slabs)
  uintptr_t base = (uintptr_t)d_ws;
  auto alloc = [&](size_t bytes) -> void* {
    void* p = (void*)base;
    base += (bytes + 255) & ~(size_t)255;
    return p;
  };
  uint16_t* wih_e  = (uint16_t*)alloc((size_t)G3H * EE * 2);
  uint16_t* whh_e  = (uint16_t*)alloc((size_t)G3H * HH * 2);
  uint16_t* wih_d  = (uint16_t*)alloc((size_t)G3H * (EE + HH) * 2);
  uint16_t* whh_d  = (uint16_t*)alloc((size_t)G3H * HH * 2);
  uint16_t* wfc    = (uint16_t*)alloc((size_t)BV  * HH * 2);
  uint16_t* ex_bf  = (uint16_t*)alloc((size_t)BB * TX * EE * 2);
  uint16_t* dec_emb= (uint16_t*)alloc((size_t)BB * TDEC * EE * 2);
  float*    eh_f32 = (float*)   alloc((size_t)BB * TX * HH * 4);
  // contiguous zero-init block: h, hx (fp32) then h_bf, hx_bf (bf16)
  float*    h_f32  = (float*)   alloc((size_t)BB * HH * 4 * 2 + (size_t)BB * HH * 2 * 2);
  float*    hx_f32 = h_f32 + BB * HH;
  uint16_t* h_bf   = (uint16_t*)(hx_f32 + BB * HH);
  uint16_t* hx_bf  = h_bf + BB * HH;
  float*    ox_f32 = (float*)   alloc((size_t)BB * HH * 4);
  float*    sx_f32 = (float*)   alloc((size_t)BB * HH * 4);
  uint16_t* dec_in = (uint16_t*)alloc((size_t)BB * (EE + HH) * 2);
  float*    gx     = (float*)   alloc((size_t)BB * G3H * 4);
  float*    gh     = (float*)   alloc((size_t)BB * G3H * 4);
  uint16_t* outs_bf= (uint16_t*)alloc((size_t)BB * TDEC * HH * 2);

  auto cvt = [&](const float* s, uint16_t* d, int n) {
    cvt_bf16_kernel<<<(n + 255) / 256, 256, 0, stream>>>(s, d, n);
  };
  cvt(enc_W_ih, wih_e, G3H * EE);
  cvt(enc_W_hh, whh_e, G3H * HH);
  cvt(dec_W_ih, wih_d, G3H * (EE + HH));
  cvt(dec_W_hh, whh_d, G3H * HH);
  cvt(fc_W,     wfc,   BV * HH);

  embed_kernel<<<(BB * TX * EE + 255) / 256, 256, 0, stream>>>(embedding, x, ex_bf, TX, TX, BB * TX * EE);
  embed_kernel<<<(BB * TDEC * EE + 255) / 256, 256, 0, stream>>>(embedding, y, dec_emb, TDEC, TY, BB * TDEC * EE);

  int zn = (BB * HH * 4 * 2 + BB * HH * 2 * 2) / 4;     // 49152 dwords: h, hx, h_bf, hx_bf
  fill_zero_kernel<<<(zn + 255) / 256, 256, 0, stream>>>((uint32_t*)h_f32, zn);
  init_ox_kernel<<<(BB * HH + 255) / 256, 256, 0, stream>>>(h_init, ox_f32);

  // gate GEMM grid: 2 * (32/16)*(1536/16) = 384 waves = 12288 threads = 48 blocks
  const int GATE_BLOCKS = (2 * (BB / 16) * (G3H / 16) * 32 + 255) / 256;
  const int UPD_BLOCKS  = (BB * HH + 255) / 256;

  // ---- encoder scan
  for (int t = 0; t < TX; ++t) {
    gates_gemm_kernel<<<GATE_BLOCKS, 256, 0, stream>>>(
        ex_bf + (size_t)t * EE, TX * EE, EE, wih_e,
        h_bf, HH, whh_e, gx, gh, BB, G3H);
    gru_update_kernel<<<UPD_BLOCKS, 256, 0, stream>>>(
        gx, gh, enc_b_ih, enc_b_hh, h_f32, h_bf,
        eh_f32 + (size_t)t * HH, (long)TX * HH,
        nullptr, nullptr, 0, nullptr);
  }

  // ---- decoder scan with attention
  for (int t = 0; t < TDEC; ++t) {
    attn_kernel<<<BB, 512, 0, stream>>>(eh_f32, ox_f32, dec_emb, dec_in, sx_f32, t);
    gates_gemm_kernel<<<GATE_BLOCKS, 256, 0, stream>>>(
        dec_in, EE + HH, EE + HH, wih_d,
        hx_bf, HH, whh_d, gx, gh, BB, G3H);
    gru_update_kernel<<<UPD_BLOCKS, 256, 0, stream>>>(
        gx, gh, dec_b_ih, dec_b_hh, hx_f32, hx_bf,
        nullptr, 0,
        sx_f32, outs_bf + (size_t)t * HH, (long)TDEC * HH,
        ox_f32);
  }

  // ---- final projection: (4064 x 512) @ (512 x 32000) + bias, straight into d_out
  // 32x64 block per wave: (4064/32)*(32000/64) = 127*500 = 63500 waves
  int waves = (BB * TDEC / 32) * (BV / 64);
  int blocks = (waves * 32 + 255) / 256;
  gemm_bias_kernel<<<blocks, 256, 0, stream>>>(outs_bf, wfc, fc_b, out, BB * TDEC, BV, HH);
}
